// SparKDensfiyBlock_2327872274478
// MI455X (gfx1250) — matrix operand order repaired and checked by reading, not compiled, not run on hardware
//
#include <hip/hip_runtime.h>

// ---------------------------------------------------------------------------
// SparK densify block (channels-first LN + mask-token fill + 3x3 SAME conv
// 512->512) as a fused implicit-GEMM on CDNA5 v_wmma_f32_16x16x32_bf16.
//   B=32, C=512, H=W=56, D=512  ->  GEMM M=512, N=100352, K=C*9=4608
//
// Roofline: 0.47 TFLOP of MACs vs >=420 MB HBM traffic -> memory bound
// (~18us floor @ 23.3 TB/s); bf16 WMMA makes math free. Secondary resource is
// L2 weight re-reads: traffic = M*K*2 * (N_total/N_blk) -> full-row N tile
// (N_blk=64) keeps it at ~8.4 GB of L2-resident reads. A-fragments are
// ping-pong double-buffered across an unroll-by-2 k-loop (no register-copy
// rotation, no WMMA->VALU hazard nops) and addressed via per-mtile base
// pointers + constant k-step offsets folded into the load IOFFSET.
// ---------------------------------------------------------------------------

typedef __attribute__((ext_vector_type(16))) __bf16 v16bf;
typedef __attribute__((ext_vector_type(8)))  float  v8f;

#define Bsz 32
#define Cch 512
#define Hh  56
#define Ww  56
#define Dd  512
#define KTILES 144      // 4608 / 32 k-tiles of the packed weight panel
#define CCHUNK 64       // channels per LDS im2col chunk
#define KCHUNK 576      // CCHUNK*9
#define KSTEPS 18       // KCHUNK/32 (even -> clean ping-pong)
#define NCHUNKS 8
#define NBLK 64         // padded N tile (56 real columns = one output row)
#define PCOLS 66        // patch columns (w -1 .. 64)
#define NPOS 198        // 3 rows x 66 cols patch positions
#define BSTRIDE 592     // halves per im2col row; 37*16 -> 32B aligned frags

__device__ inline __bf16 f2bf(float f) {
  unsigned u = __builtin_bit_cast(unsigned, f);
  u += 0x7fffu + ((u >> 16) & 1u);            // round-to-nearest-even
  unsigned short s = (unsigned short)(u >> 16);
  return __builtin_bit_cast(__bf16, s);
}

// Repack conv_w (D,C,3,3) f32 -> bf16 A-fragments in per-lane WMMA layout:
// pw[((mt*KTILES + kt)*32 + lane)*16 + i]  (one contiguous 32B chunk / lane).
// contiguous k order used here (B matches; not AMD's table order): lanes 0-15 rows M=0-15 holding K {0-15},
// lanes 16-31 same rows holding K {16-31}.
__global__ void spark_pack_w(const float* __restrict__ w, __bf16* __restrict__ pw) {
  int t = blockIdx.x * blockDim.x + threadIdx.x;
  if (t >= 32 * KTILES * 32) return;
  int lane = t & 31;
  int kt   = (t >> 5) % KTILES;
  int mt   = t / (32 * KTILES);
  int d    = mt * 16 + (lane & 15);
  int hi   = lane >> 4;
  v16bf vv;
#pragma unroll
  for (int i = 0; i < 16; ++i) {
    int g = i >> 3, j = i & 7;
    int k = kt * 32 + hi * 16 + g * 8 + j;    // k = c*9 + kh*3 + kw
    int c = k / 9;
    int rem = k % 9;
    int kh = rem / 3, kw = rem % 3;
    vv[i] = f2bf(w[((d * Cch + c) * 3 + kh) * 3 + kw]);
  }
  ((v16bf*)pw)[t] = vv;
}

__global__ void __launch_bounds__(256) spark_densify(
    const float* __restrict__ bcff,
    const unsigned char* __restrict__ act,   // jax bool -> 1 byte
    const float* __restrict__ mtok,
    const float* __restrict__ gamma,
    const float* __restrict__ beta,
    const float* __restrict__ convb,
    const __bf16* __restrict__ pw,
    float* __restrict__ out)
{
  const int h    = blockIdx.x;       // 0..55
  const int b    = blockIdx.y;       // 0..31
  const int tid  = threadIdx.x;
  const int lane = tid & 31;
  const int wave = tid >> 5;

  __shared__ __align__(32) __bf16 Bs[NBLK][BSTRIDE];   // 75,776 B im2col panel
  __shared__ float mu_s[NPOS], rs_s[NPOS];
  __shared__ unsigned char act_s[NPOS + 2];

  // ---- Phase A: LN stats, one thread per patch position (3 x 66) ----------
  if (tid < NPOS) {
    int row = tid / PCOLS, col = tid % PCOLS;
    int ra = h - 1 + row, ca = col - 1;
    float s = 0.f, sq = 0.f;
    unsigned char a = 0;
    if (ra >= 0 && ra < Hh && ca >= 0 && ca < Ww) {
      const float* ptr = bcff + ((b * Cch) * Hh + ra) * Ww + ca;
      for (int c = 0; c < Cch; ++c) {
        float v = ptr[c * (Hh * Ww)];
        s += v; sq += v * v;
      }
      a = act[(b * Hh + ra) * Ww + ca] ? 1 : 0;
    }
    float mu  = s * (1.0f / Cch);
    float var = sq * (1.0f / Cch) - mu * mu;
    mu_s[tid]  = mu;
    rs_s[tid]  = rsqrtf(var + 1e-6f);
    act_s[tid] = a;
  }
  __syncthreads();

  v8f acc[4][4] = {};                // [mtile i][ntile j]
  const v16bf* aG = (const v16bf*)pw;
  const int nb    = lane & 15;
  const int khalf = (lane >> 4) * 16;

  // ---- Phase B: per 64-channel chunk: normalize+im2col into LDS, WMMA -----
  for (int cc = 0; cc < NCHUNKS; ++cc) {
    unsigned* bz = (unsigned*)&Bs[0][0];
    for (int i = tid; i < NBLK * BSTRIDE / 2; i += 256) bz[i] = 0u;
    __syncthreads();

    // warm next chunk's weight panel while we do im2col (global_prefetch_b8)
    if (cc + 1 < NCHUNKS)
      __builtin_prefetch((const void*)&aG[((wave * 4) * KTILES +
                                          (cc + 1) * KSTEPS) * 32 + lane], 0, 1);

    for (int idx = tid; idx < CCHUNK * NPOS; idx += 256) {
      int cl = idx / NPOS;
      int p  = idx % NPOS;
      int row = p / PCOLS, col = p % PCOLS;
      int ra = h - 1 + row, ca = col - 1;
      if (ra < 0 || ra >= Hh || ca < 0 || ca >= Ww) continue;  // SAME pad = 0
      int c = cc * CCHUNK + cl;
      float v = bcff[((b * Cch + c) * Hh + ra) * Ww + ca];
      float xn = act_s[p] ? (v - mu_s[p]) * rs_s[p] * gamma[c] + beta[c]
                          : mtok[c];
      __bf16 hv = f2bf(xn);
#pragma unroll
      for (int kw = 0; kw < 3; ++kw) {
        int n = col - kw;                 // output column this tap feeds
        if (n >= 0 && n < NBLK)
          Bs[n][cl * 9 + row * 3 + kw] = hv;
      }
    }
    __syncthreads();

    // ---- WMMA: 18 k-steps, 4 mtiles x 4 ntiles, ping-pong A buffers ------
    // Per-mtile base pointers: k-step stride is a constant 32 v16bf = 1024 B,
    // so pAi[ks] folds into the load's immediate offset (no per-step VALU).
    const int ktbase = cc * KSTEPS;
    const v16bf* pA0 = aG + ((wave * 4 + 0) * KTILES + ktbase) * 32 + lane;
    const v16bf* pA1 = aG + ((wave * 4 + 1) * KTILES + ktbase) * 32 + lane;
    const v16bf* pA2 = aG + ((wave * 4 + 2) * KTILES + ktbase) * 32 + lane;
    const v16bf* pA3 = aG + ((wave * 4 + 3) * KTILES + ktbase) * 32 + lane;

    v16bf a[4], an[4];
    a[0] = pA0[0]; a[1] = pA1[0]; a[2] = pA2[0]; a[3] = pA3[0];

#pragma unroll
    for (int ks = 0; ks < KSTEPS; ks += 2) {
      // ---- even step: consume a[], prefetch an[] = k-step ks+1 -----------
      {
        v16bf bf[4];
#pragma unroll
        for (int j = 0; j < 4; ++j)
          bf[j] = *(const v16bf*)&Bs[j * 16 + nb][ks * 32 + khalf];
        an[0] = pA0[(ks + 1) * 32];
        an[1] = pA1[(ks + 1) * 32];
        an[2] = pA2[(ks + 1) * 32];
        an[3] = pA3[(ks + 1) * 32];
#pragma unroll
        for (int i = 0; i < 4; ++i)
#pragma unroll
          for (int j = 0; j < 4; ++j)
            acc[i][j] = __builtin_amdgcn_wmma_f32_16x16x32_bf16(
                false, a[i], false, bf[j], (short)0, acc[i][j], false, false);
      }
      // ---- odd step: consume an[], prefetch a[] = k-step ks+2 ------------
      {
        v16bf bf[4];
#pragma unroll
        for (int j = 0; j < 4; ++j)
          bf[j] = *(const v16bf*)&Bs[j * 16 + nb][(ks + 1) * 32 + khalf];
        if (ks + 2 < KSTEPS) {
          a[0] = pA0[(ks + 2) * 32];
          a[1] = pA1[(ks + 2) * 32];
          a[2] = pA2[(ks + 2) * 32];
          a[3] = pA3[(ks + 2) * 32];
        }
#pragma unroll
        for (int i = 0; i < 4; ++i)
#pragma unroll
          for (int j = 0; j < 4; ++j)
            acc[i][j] = __builtin_amdgcn_wmma_f32_16x16x32_bf16(
                false, an[i], false, bf[j], (short)0, acc[i][j], false, false);
      }
    }
    __syncthreads();
  }

  // ---- Epilogue: + conv_b, store one full (B,D,H,W) f32 row ---------------
  const int wl   = lane & 15;
  const int mrow = 8 * (lane >> 4);
#pragma unroll
  for (int j = 0; j < 4; ++j) {
    int w = j * 16 + wl;
    if (w >= Ww) continue;              // ragged last ntile (56 = 3*16+8)
#pragma unroll
    for (int i = 0; i < 4; ++i) {
      int d0 = (wave * 4 + i) * 16 + mrow;
#pragma unroll
      for (int r = 0; r < 8; ++r) {
        int d = d0 + r;
        out[((b * Dd + d) * Hh + h) * Ww + w] = acc[i][j][r] + convb[d];
      }
    }
  }
}

extern "C" void kernel_launch(void* const* d_in, const int* in_sizes, int n_in,
                              void* d_out, int out_size, void* d_ws, size_t ws_size,
                              hipStream_t stream) {
  const float*         bcff  = (const float*)d_in[0];
  const unsigned char* activ = (const unsigned char*)d_in[1]; // bool array
  const float*         mtok  = (const float*)d_in[2];
  const float*         gam   = (const float*)d_in[3];
  const float*         bet   = (const float*)d_in[4];
  const float*         convw = (const float*)d_in[5];
  const float*         convb = (const float*)d_in[6];
  float*               outp  = (float*)d_out;
  __bf16*              pw    = (__bf16*)d_ws;   // 512*4608*2B = 4.5 MB

  int nPack = 32 * KTILES * 32;
  spark_pack_w<<<(nPack + 255) / 256, 256, 0, stream>>>(convw, pw);

  dim3 grid(Hh, Bsz);                  // one block per (row, batch)
  spark_densify<<<grid, 256, 0, stream>>>(bcff, activ, mtok, gam, bet, convb,
                                          pw, outp);
}
